// GPT_12524124635720
// MI455X (gfx1250) — matrix operand order repaired and checked by reading, not compiled, not run on hardware
//
#include <hip/hip_runtime.h>
#include <hip/hip_fp16.h>

// ---------------- model dims (fixed by reference) ----------------
#define GPT_V  32000
#define GPT_L  6
#define GPT_H  12
#define GPT_D  768
#define GPT_T  1024
#define GPT_B  4
#define GPT_HD 64
#define GPT_FF 3072
#define ROWS   (GPT_B * GPT_T)   // 4096 token rows

// ---------------- GEMM tiling ----------------
#define TM   128      // block tile M
#define TN   128      // block tile N
#define KS   32       // K per pipeline round (one WMMA K-chunk)
#define LDSS 40       // padded LDS row stride in halfs (80B, 16B aligned)

typedef __attribute__((ext_vector_type(16))) _Float16 v16h;
typedef __attribute__((ext_vector_type(8)))  _Float16 v8h;
typedef __attribute__((ext_vector_type(8)))  float    v8f;

union Frag { v8h h[2]; v16h v; };

// -------- CDNA5 async global->LDS copy (no VGPR round-trip, ASYNCcnt) -------
// ROCm clang-22 signature: (AS1 int-vec4*, AS3 int-vec4*, imm offset, imm cpol)
#if defined(__has_builtin)
#if __has_builtin(__builtin_amdgcn_global_load_async_to_lds_b128)
#define HAVE_ASYNC_LDS 1
#endif
#endif

#ifdef HAVE_ASYNC_LDS
typedef int gv4i __attribute__((vector_size(16)));
#define ASYNC_CP16(g, l, off)                                                  \
    __builtin_amdgcn_global_load_async_to_lds_b128(                            \
        (__attribute__((address_space(1))) gv4i*)(uintptr_t)(g),               \
        (__attribute__((address_space(3))) gv4i*)(unsigned)(uintptr_t)(l),     \
        (off), 0)
#endif

// stage one 32B slice of A and B tiles (per thread) into LDS
static __device__ __forceinline__ void stage_tiles(const _Float16* ga,
                                                   const _Float16* gb,
                                                   _Float16* la, _Float16* lb)
{
#ifdef HAVE_ASYNC_LDS
    ASYNC_CP16(ga, la, 0);  ASYNC_CP16(ga, la, 16);
    ASYNC_CP16(gb, lb, 0);  ASYNC_CP16(gb, lb, 16);
#else
    v8h ta0 = ((const v8h*)ga)[0], ta1 = ((const v8h*)ga)[1];
    v8h tb0 = ((const v8h*)gb)[0], tb1 = ((const v8h*)gb)[1];
    ((v8h*)la)[0] = ta0; ((v8h*)la)[1] = ta1;
    ((v8h*)lb)[0] = tb0; ((v8h*)lb)[1] = tb1;
#endif
}

// =====================================================================
// Embedding: x[b,t,:] = tok_emb[idx[b,t]] + pos_emb[t]   (f32)
// =====================================================================
__global__ __launch_bounds__(256)
void embed_kernel(const int* __restrict__ idx, const float* __restrict__ tok,
                  const float* __restrict__ pos, float* __restrict__ x)
{
    int row = blockIdx.x;                 // b*T + t
    int t   = row & (GPT_T - 1);
    int id  = idx[row];
    id = id < 0 ? 0 : (id >= GPT_V ? GPT_V - 1 : id);
    const float* tr = tok + (size_t)id * GPT_D;
    const float* pr = pos + (size_t)t  * GPT_D;
    float* xr = x + (size_t)row * GPT_D;
    for (int d = threadIdx.x; d < GPT_D; d += 256)
        xr[d] = tr[d] + pr[d];
}

// =====================================================================
// Tiled transpose + f32->f16 convert: out[z][n][k] = (f16) in[z][k][n]
// grid = (K/32, N/32, nmat), block = (32,8). Coalesced on both sides.
// =====================================================================
__global__ __launch_bounds__(256)
void transpose_cvt_kernel(const float* __restrict__ in, _Float16* __restrict__ out,
                          int K, int N)
{
    __shared__ float t[32][33];
    const int k0 = blockIdx.x * 32;
    const int n0 = blockIdx.y * 32;
    const float*    src = in  + (size_t)blockIdx.z * K * N;
    _Float16*       dst = out + (size_t)blockIdx.z * K * N;
    const int tx = threadIdx.x, ty = threadIdx.y;
#pragma unroll
    for (int i = 0; i < 4; ++i)
        t[ty + 8 * i][tx] = src[(size_t)(k0 + ty + 8 * i) * N + n0 + tx];
    __syncthreads();
#pragma unroll
    for (int i = 0; i < 4; ++i)
        dst[(size_t)(n0 + ty + 8 * i) * K + k0 + tx] = (_Float16)t[tx][ty + 8 * i];
}

// =====================================================================
// LayerNorm over D=768 per row; emits f16 (only GEMM-A consumes it)
// =====================================================================
__global__ __launch_bounds__(256)
void layernorm_kernel(const float* __restrict__ x, const float* __restrict__ w,
                      const float* __restrict__ b, _Float16* __restrict__ out)
{
    __shared__ float s1[256], s2[256];
    int row = blockIdx.x, tid = threadIdx.x;
    const float* xr = x + (size_t)row * GPT_D;
    float sum = 0.f, sq = 0.f;
    for (int d = tid; d < GPT_D; d += 256) { float v = xr[d]; sum += v; sq += v * v; }
    s1[tid] = sum; s2[tid] = sq;
    __syncthreads();
    for (int o = 128; o > 0; o >>= 1) {
        if (tid < o) { s1[tid] += s1[tid + o]; s2[tid] += s2[tid + o]; }
        __syncthreads();
    }
    float mean = s1[0] * (1.0f / GPT_D);
    float var  = s2[0] * (1.0f / GPT_D) - mean * mean;
    float inv  = rsqrtf(var + 1e-5f);
    _Float16* orow = out + (size_t)row * GPT_D;
    for (int d = tid; d < GPT_D; d += 256)
        orow[d] = (_Float16)((xr[d] - mean) * inv * w[d] + b[d]);
}

// =====================================================================
// WMMA GEMM, all-f16 operands, f32 accumulate:
//   C[M,N] = act( A[M,K](f16) * BT[N,K]^T(f16) + bias + residual )
//   Block 256 thr (8 waves), 128x128 C tile; wave = 32x64 (2x4 WMMA frags).
//   Double-buffered async pipeline: round r+1's global->LDS DMA
//   (global_load_async_to_lds_b128, ASYNCcnt) overlaps round r's WMMAs.
//   One barrier per round.
// =====================================================================
__global__ __launch_bounds__(256)
void gemm_f16_wmma(const _Float16* __restrict__ A, const _Float16* __restrict__ BT,
                   const float* __restrict__ bias, const float* __restrict__ residual,
                   void* __restrict__ C, int M, int N, int K, int c_f16, int relu)
{
    __shared__ __align__(16) _Float16 sA[2][TM * LDSS];   // [buf][m][k]
    __shared__ __align__(16) _Float16 sB[2][TN * LDSS];   // [buf][n][k]

    const int tid  = threadIdx.x;
    const int n0   = blockIdx.x * TN;
    const int m0   = blockIdx.y * TM;
    const int lane = tid & 31;
    const int wave = tid >> 5;
    const int wr = wave >> 1;          // 0..3 : 32-row strip
    const int wc = wave & 1;           // 0..1 : 64-col strip
    const int lane16 = lane & 15;
    const int half   = lane >> 4;

    v8f acc[2][4];
    const v8f vzero = {0.f,0.f,0.f,0.f,0.f,0.f,0.f,0.f};
#pragma unroll
    for (int i = 0; i < 2; ++i)
#pragma unroll
        for (int j = 0; j < 4; ++j) acc[i][j] = vzero;

    // staging: each thread copies 32B (2x b128) for A and for B per round
    const int sr  = tid >> 1;          // 0..127 (row for A / n for B)
    const int sc0 = (tid & 1) << 4;    // 0 or 16 halfs

    const _Float16* gA = A  + (size_t)(m0 + sr) * K + sc0;
    const _Float16* gB = BT + (size_t)(n0 + sr) * K + sc0;
    const int soff = sr * LDSS + sc0;

    const int nrounds = K / KS;

    // prologue: stage round 0 into buffer 0
    stage_tiles(gA, gB, &sA[0][soff], &sB[0][soff]);

    for (int r = 0; r < nrounds; ++r) {
#ifdef HAVE_ASYNC_LDS
        asm volatile("s_wait_asynccnt 0" ::: "memory");
#endif
        __syncthreads();   // round-r tiles visible; buffer (r+1)&1 free

        if (r + 1 < nrounds) {
            const int nb = (r + 1) & 1;
            stage_tiles(gA + (size_t)(r + 1) * KS, gB + (size_t)(r + 1) * KS,
                        &sA[nb][soff], &sB[nb][soff]);
            if (r + 2 < nrounds) {
                __builtin_prefetch(gA + (size_t)(r + 2) * KS, 0, 1);
                __builtin_prefetch(gB + (size_t)(r + 2) * KS, 0, 1);
            }
        }

        // ---- compute round r from buffer r&1 (overlaps round r+1 DMA) ----
        const _Float16* cA = sA[r & 1];
        const _Float16* cB = sB[r & 1];
        Frag a[2], b[4];
        // A frag (16-bit A 16x32): lanes 0-15 K{0-7,16-23}, lanes 16-31 K{8-15,24-31}
#pragma unroll
        for (int mi = 0; mi < 2; ++mi) {
            int row = wr * 32 + mi * 16 + lane16;
            a[mi].h[0] = *(const v8h*)&cA[row * LDSS + half * 8];
            a[mi].h[1] = *(const v8h*)&cA[row * LDSS + 16 + half * 8];
        }
        // B frag (16-bit B 32x16): lanes 0-15 K{0-7,16-23}, lanes 16-31 K{8-15,24-31}
#pragma unroll
        for (int ni = 0; ni < 4; ++ni) {
            int col = wc * 64 + ni * 16 + lane16;
            b[ni].h[0] = *(const v8h*)&cB[col * LDSS + half * 8];
            b[ni].h[1] = *(const v8h*)&cB[col * LDSS + 16 + half * 8];
        }
#pragma unroll
        for (int mi = 0; mi < 2; ++mi)
#pragma unroll
            for (int ni = 0; ni < 4; ++ni)
                acc[mi][ni] = __builtin_amdgcn_wmma_f32_16x16x32_f16(
                    false, a[mi].v, false, b[ni].v, (short)0, acc[mi][ni],
                    false, false);
    }

    // Epilogue. C/D map: VGPR j -> M = j + 8*half, N = lane16.
#pragma unroll
    for (int mi = 0; mi < 2; ++mi) {
#pragma unroll
        for (int ni = 0; ni < 4; ++ni) {
            const int col = n0 + wc * 64 + ni * 16 + lane16;
            const int r0  = m0 + wr * 32 + mi * 16 + half * 8;
            const float bv = bias ? bias[col] : 0.0f;
            size_t base = (size_t)r0 * N + col;
#pragma unroll
            for (int j = 0; j < 8; ++j, base += N) {
                float vo = acc[mi][ni][j] + bv;
                if (residual) vo += residual[base];
                if (relu) vo = fmaxf(vo, 0.0f);
                if (c_f16) ((_Float16*)C)[base] = (_Float16)vo;
                else       ((float*)   C)[base] = vo;
            }
        }
    }
}

// =====================================================================
// Causal attention, streaming online softmax. One wave per (b,h,t) row.
// q,k,v f32 [B,T,H,64]; o written f16 (feeds GEMM A). scale = D^-0.5.
// =====================================================================
__global__ __launch_bounds__(256)
void attention_kernel(const float* __restrict__ q, const float* __restrict__ k,
                      const float* __restrict__ v, _Float16* __restrict__ o)
{
    const int w    = blockIdx.x * 8 + (threadIdx.x >> 5);
    const int lane = threadIdx.x & 31;
    const int t  = w & (GPT_T - 1);
    const int bh = w >> 10;
    const int h  = bh % GPT_H;
    const int b  = bh / GPT_H;

    const float scale = 0.0360843918243516f; // 768^-0.5 (reference quirk)
    const size_t rowstride = (size_t)GPT_H * GPT_HD; // 768

    const float* qr = q + ((size_t)(b * GPT_T + t) * GPT_H + h) * GPT_HD;
    float q0 = qr[lane], q1 = qr[lane + 32];

    float m = -1e30f, l = 0.f, acc0 = 0.f, acc1 = 0.f;
    const float* kr = k + ((size_t)b * GPT_T * GPT_H + h) * GPT_HD;
    const float* vr = v + ((size_t)b * GPT_T * GPT_H + h) * GPT_HD;

    for (int s = 0; s <= t; ++s) {
        const float* kp = kr + (size_t)s * rowstride;
        float p = q0 * kp[lane] + q1 * kp[lane + 32];
        p += __shfl_xor(p, 16);
        p += __shfl_xor(p, 8);
        p += __shfl_xor(p, 4);
        p += __shfl_xor(p, 2);
        p += __shfl_xor(p, 1);
        float sc   = p * scale;
        float mn   = fmaxf(m, sc);
        float corr = __expf(m - mn);
        float pe   = __expf(sc - mn);
        const float* vp = vr + (size_t)s * rowstride;
        l    = l * corr + pe;
        acc0 = acc0 * corr + pe * vp[lane];
        acc1 = acc1 * corr + pe * vp[lane + 32];
        m = mn;
    }
    float inv = 1.0f / l;
    _Float16* orow = o + ((size_t)(b * GPT_T + t) * GPT_H + h) * GPT_HD;
    orow[lane]      = (_Float16)(acc0 * inv);
    orow[lane + 32] = (_Float16)(acc1 * inv);
}

// =====================================================================
// Host-side orchestration (graph-capture safe: launches on stream only)
// =====================================================================
extern "C" void kernel_launch(void* const* d_in, const int* in_sizes, int n_in,
                              void* d_out, int out_size, void* d_ws, size_t ws_size,
                              hipStream_t stream)
{
    (void)in_sizes; (void)n_in; (void)out_size; (void)ws_size;

    const int*   idx     = (const int*)  d_in[0];
    const float* tok_emb = (const float*)d_in[1];
    const float* pos_emb = (const float*)d_in[2];
    const float* Wq      = (const float*)d_in[3];
    const float* Wk      = (const float*)d_in[4];
    const float* Wv      = (const float*)d_in[5];
    const float* Wo      = (const float*)d_in[6];
    const float* bo      = (const float*)d_in[7];
    const float* ln1_w   = (const float*)d_in[8];
    const float* ln1_b   = (const float*)d_in[9];
    const float* W1      = (const float*)d_in[10];
    const float* b1      = (const float*)d_in[11];
    const float* W2      = (const float*)d_in[12];
    const float* b2      = (const float*)d_in[13];
    const float* ln2_w   = (const float*)d_in[14];
    const float* ln2_b   = (const float*)d_in[15];
    const float* lnf_w   = (const float*)d_in[16];
    const float* lnf_b   = (const float*)d_in[17];
    const float* Wout    = (const float*)d_in[18];
    const float* bout    = (const float*)d_in[19];

    float* out = (float*)d_out;

    // ---------------- workspace layout ----------------
    const size_t RD  = (size_t)ROWS * GPT_D;         // 4096*768
    const size_t RF  = (size_t)ROWS * GPT_FF;        // 4096*3072
    const size_t WDD = (size_t)GPT_D * GPT_D;        // 589824
    const size_t WDF = (size_t)GPT_D * GPT_FF;       // 2359296

    float*    x   = (float*)d_ws;        // residual stream, f32
    float*    qb  = x  + RD;
    float*    kb  = qb + RD;
    float*    vb  = kb + RD;
    _Float16* hb  = (_Float16*)(vb + RD);            // LN out (GEMM A)
    _Float16* ab  = hb + RD;                         // attn out (GEMM A)
    _Float16* fb  = ab + RD;                         // relu out (GEMM A)
    _Float16* wq16   = fb + RF;                      // f16 transposed weights
    _Float16* wk16   = wq16 + GPT_L * WDD;
    _Float16* wv16   = wk16 + GPT_L * WDD;
    _Float16* wo16   = wv16 + GPT_L * WDD;
    _Float16* w116   = wo16 + GPT_L * WDD;           // [N=3072][K=768]
    _Float16* w216   = w116 + GPT_L * WDF;           // [N=768][K=3072]
    _Float16* wout16 = w216 + GPT_L * WDF;           // [N=32000][K=768]

    const dim3 blk(256);
    const dim3 tblk(32, 8);

    // ---- per-launch weight convert+transpose: f32[K][N] -> f16[N][K]
    transpose_cvt_kernel<<<dim3(GPT_D/32, GPT_HD/32, GPT_L*GPT_H), tblk, 0, stream>>>(Wq, wq16, GPT_D, GPT_HD);
    transpose_cvt_kernel<<<dim3(GPT_D/32, GPT_HD/32, GPT_L*GPT_H), tblk, 0, stream>>>(Wk, wk16, GPT_D, GPT_HD);
    transpose_cvt_kernel<<<dim3(GPT_D/32, GPT_HD/32, GPT_L*GPT_H), tblk, 0, stream>>>(Wv, wv16, GPT_D, GPT_HD);
    transpose_cvt_kernel<<<dim3(GPT_D/32, GPT_D/32,  GPT_L),       tblk, 0, stream>>>(Wo, wo16, GPT_D, GPT_D);
    transpose_cvt_kernel<<<dim3(GPT_D/32, GPT_FF/32, GPT_L),       tblk, 0, stream>>>(W1, w116, GPT_D, GPT_FF);
    transpose_cvt_kernel<<<dim3(GPT_FF/32, GPT_D/32, GPT_L),       tblk, 0, stream>>>(W2, w216, GPT_FF, GPT_D);
    transpose_cvt_kernel<<<dim3(GPT_D/32, GPT_V/32,  1),           tblk, 0, stream>>>(Wout, wout16, GPT_D, GPT_V);

    embed_kernel<<<ROWS, blk, 0, stream>>>(idx, tok_emb, pos_emb, x);

    const dim3 gD (GPT_D  / TN, ROWS / TM);   // 6 x 32
    const dim3 gFF(GPT_FF / TN, ROWS / TM);   // 24 x 32
    const dim3 gV (GPT_V  / TN, ROWS / TM);   // 250 x 32

    for (int l = 0; l < GPT_L; ++l) {
        const _Float16* wq_l = wq16 + (size_t)l * WDD;
        const _Float16* wk_l = wk16 + (size_t)l * WDD;
        const _Float16* wv_l = wv16 + (size_t)l * WDD;
        const _Float16* wo_l = wo16 + (size_t)l * WDD;
        const _Float16* w1_l = w116 + (size_t)l * WDF;
        const _Float16* w2_l = w216 + (size_t)l * WDF;

        layernorm_kernel<<<ROWS, blk, 0, stream>>>(x, ln1_w + l * GPT_D,
                                                   ln1_b + l * GPT_D, hb);
        gemm_f16_wmma<<<gD, blk, 0, stream>>>(hb, wq_l, nullptr, nullptr, qb,
                                              ROWS, GPT_D, GPT_D, 0, 0);
        gemm_f16_wmma<<<gD, blk, 0, stream>>>(hb, wk_l, nullptr, nullptr, kb,
                                              ROWS, GPT_D, GPT_D, 0, 0);
        gemm_f16_wmma<<<gD, blk, 0, stream>>>(hb, wv_l, nullptr, nullptr, vb,
                                              ROWS, GPT_D, GPT_D, 0, 0);
        attention_kernel<<<(GPT_B * GPT_H * GPT_T) / 8, blk, 0, stream>>>(qb, kb, vb, ab);
        gemm_f16_wmma<<<gD, blk, 0, stream>>>(ab, wo_l, bo + l * GPT_D, x, x,
                                              ROWS, GPT_D, GPT_D, 0, 0);
        layernorm_kernel<<<ROWS, blk, 0, stream>>>(x, ln2_w + l * GPT_D,
                                                   ln2_b + l * GPT_D, hb);
        gemm_f16_wmma<<<gFF, blk, 0, stream>>>(hb, w1_l, b1 + l * GPT_FF, nullptr,
                                               fb, ROWS, GPT_FF, GPT_D, 1, 1);
        gemm_f16_wmma<<<gD, blk, 0, stream>>>(fb, w2_l, b2 + l * GPT_D, x, x,
                                              ROWS, GPT_D, GPT_FF, 0, 0);
    }

    layernorm_kernel<<<ROWS, blk, 0, stream>>>(x, lnf_w, lnf_b, hb);
    gemm_f16_wmma<<<gV, blk, 0, stream>>>(hb, wout16, bout, nullptr, out,
                                          ROWS, GPT_V, GPT_D, 0, 0);
}
